// EagerLSTM_48481590837457
// MI455X (gfx1250) — matrix operand order repaired and checked by reading, not compiled, not run on hardware
//
#include <hip/hip_runtime.h>

// ---------------------------------------------------------------------------
// EagerLSTM on MI455X (gfx1250, wave32, WMMA bf16 + Tensor Data Mover)
//   B=64, T=512, D=1024, U=1024
// Phase 1: convert x -> bf16, Wx/Wh -> bf16 transposed ([N][K], K-contiguous)
// Phase 2: x_proj = x @ Wx  (WMMA bf16 GEMM, TDM async tile staging into LDS,
//          double buffered, nontemporal fp32 output stream)
// Phase 3: persistent recurrent kernel, 64 WGs, grid barrier per timestep,
//          Wh L2-resident, c-state LDS-resident, h republished in bf16.
// ---------------------------------------------------------------------------

#define B_    64
#define T_    512
#define D_    1024
#define U_    1024
#define FOURU 4096
#define BT    (B_ * T_)

typedef __bf16 bf16;
typedef __attribute__((ext_vector_type(16))) __bf16 v16bf;
typedef __attribute__((ext_vector_type(8)))  float  v8f;
typedef __attribute__((ext_vector_type(4)))  unsigned int u32v4;
typedef __attribute__((ext_vector_type(8)))  int          i32v8;
typedef __attribute__((ext_vector_type(4)))  int          i32v4;

static __device__ __forceinline__ v8f zero8() {
    v8f z;
    for (int i = 0; i < 8; ++i) z[i] = 0.0f;
    return z;
}

// ---------------------------------------------------------------------------
// Tensor Data Mover: async 2D tile load (bf16) global -> LDS.
// Tile: tile_rows x tile_k elements, global row stride = stride_elems,
// packed contiguously into LDS at lds_byte_off.
// Descriptor per CDNA5 ISA ch.8 (groups 0/1; groups 2/3 zero for 2D).
// This toolchain exposes the 6-arg builtin:
//   (uint32x4 g0, int32x8 g1, int32x4 g2, int32x4 g3, int32x8 gx, int cpol)
// ---------------------------------------------------------------------------
static __device__ __forceinline__ void tdm_load_tile_bf16(
    const bf16* gptr, unsigned int lds_byte_off,
    unsigned int tile_k, unsigned int tile_rows, unsigned int stride_elems)
{
    const unsigned long long ga = (unsigned long long)(size_t)gptr;
    const unsigned int dim1 = 1u << 20;  // generous: no OOB clipping

    u32v4 g0;
    g0[0] = 1u;                                         // count=1, user desc
    g0[1] = lds_byte_off;                               // LDS dest (bytes)
    g0[2] = (unsigned int)(ga & 0xFFFFFFFFull);         // global addr [31:0]
    g0[3] = (unsigned int)((ga >> 32) & 0x1FFFFFFu)     // global addr [56:32]
          | (2u << 30);                                 // type = 2 ("image")

    i32v8 g1;
    g1[0] = 0x00010000;                                 // data_size=1 (2 bytes)
    g1[1] = (int)((stride_elems & 0xFFFFu) << 16);      // tensor_dim0 [15:0]
    g1[2] = (int)((stride_elems >> 16) | ((dim1 & 0xFFFFu) << 16));
    g1[3] = (int)((dim1 >> 16) | (tile_k << 16));       // tile_dim0
    g1[4] = (int)tile_rows;                             // tile_dim1 (tile_dim2=0)
    g1[5] = (int)stride_elems;                          // tensor_dim0_stride lo32
    g1[6] = 0;
    g1[7] = 0;

    const i32v4 gz4 = {0, 0, 0, 0};
    const i32v8 gz8 = {0, 0, 0, 0, 0, 0, 0, 0};
    __builtin_amdgcn_tensor_load_to_lds(g0, g1, gz4, gz4, gz8, 0);
}

// ---------------------------------------------------------------------------
// Elementwise fp32 -> bf16 conversion
// ---------------------------------------------------------------------------
__global__ __launch_bounds__(256) void f32_to_bf16_kernel(
    const float* __restrict__ in, bf16* __restrict__ out, int n)
{
    int i = blockIdx.x * blockDim.x + threadIdx.x;
    if (i < n) out[i] = (bf16)in[i];
}

// ---------------------------------------------------------------------------
// Transpose fp32 [K][N] -> bf16 [N][K]  (K-contiguous for WMMA B fragments)
// ---------------------------------------------------------------------------
__global__ __launch_bounds__(256) void transpose_to_bf16_kernel(
    const float* __restrict__ in, bf16* __restrict__ out, int K, int N)
{
    __shared__ __align__(16) float tile[32][33];
    const int n0 = blockIdx.x * 32;
    const int k0 = blockIdx.y * 32;
    const int tx = threadIdx.x & 31;
    const int ty = threadIdx.x >> 5;     // 8 rows of 32 threads
    for (int r = ty; r < 32; r += 8)
        tile[r][tx] = in[(size_t)(k0 + r) * N + n0 + tx];
    __syncthreads();
    for (int r = ty; r < 32; r += 8)
        out[(size_t)(n0 + r) * K + k0 + tx] = (bf16)tile[tx][r];
}

// ---------------------------------------------------------------------------
// Zero h0 (bf16) and the grid-barrier counter
// ---------------------------------------------------------------------------
__global__ __launch_bounds__(256) void init_state_kernel(
    bf16* __restrict__ h, unsigned int* __restrict__ ctr, int n)
{
    int i = blockIdx.x * blockDim.x + threadIdx.x;
    if (i < n) h[i] = (bf16)0.0f;
    if (i == 0) *ctr = 0u;
}

// ---------------------------------------------------------------------------
// x_proj = x_bf16 [BT][D] @ Wxt [4096][1024]^T -> fp32 [BT][4096]
// 128x128 tile per 256-thread block, 8 waves, wave = one 16x128 strip.
// Tile staging by the Tensor Data Mover (wave 0 issues, all waves compute).
// ---------------------------------------------------------------------------
__global__ __launch_bounds__(256) void xproj_gemm_kernel(
    const bf16* __restrict__ A,    // [BT][D]   row-major
    const bf16* __restrict__ Bt,   // [4096][1024] (Wx transposed)
    float* __restrict__ C)         // [BT][4096]
{
    __shared__ __align__(16) bf16 As[2][128][32];
    __shared__ __align__(16) bf16 Bs[2][128][32];

    const int wave  = threadIdx.x >> 5;
    const int lane  = threadIdx.x & 31;
    const bool issuer = (wave == 0);          // TDM is per-wave, EXEC-ignored
    const size_t arow0 = (size_t)blockIdx.y * 128;
    const size_t bcol0 = (size_t)blockIdx.x * 128;

    const unsigned int asBase = (unsigned int)(size_t)&As[0][0][0];
    const unsigned int bsBase = (unsigned int)(size_t)&Bs[0][0][0];

    v8f acc[8];
    for (int j = 0; j < 8; ++j) acc[j] = zero8();

    // prologue: stage k-chunk 0 into buffer 0
    if (issuer) {
        tdm_load_tile_bf16(A  + arow0 * D_, asBase, 32u, 128u, (unsigned)D_);
        tdm_load_tile_bf16(Bt + bcol0 * D_, bsBase, 32u, 128u, (unsigned)D_);
        __builtin_amdgcn_s_wait_tensorcnt(0);
    }
    __syncthreads();

    int buf = 0;
    for (int kk = 0; kk < D_; kk += 32) {
        const bool more = (kk + 32 < D_);
        if (issuer && more) {   // async-stage next chunk while computing
            const unsigned int off = (unsigned)((buf ^ 1) * 128 * 32 * (int)sizeof(bf16));
            tdm_load_tile_bf16(A  + arow0 * D_ + kk + 32, asBase + off, 32u, 128u, (unsigned)D_);
            tdm_load_tile_bf16(Bt + bcol0 * D_ + kk + 32, bsBase + off, 32u, 128u, (unsigned)D_);
        }

        // A fragment: 16x32 bf16. lane<16: row=lane, K {0..7,16..23};
        //             lane>=16: row=lane-16, K {8..15,24..31}.
        v16bf afrag;
        {
            const int row = (wave << 4) + (lane & 15);
            const int c8  = (lane < 16) ? 0 : 8;
            *(u32v4*)&afrag       = *(const u32v4*)&As[buf][row][c8];
            *((u32v4*)&afrag + 1) = *(const u32v4*)&As[buf][row][c8 + 16];
        }

        // batch all 8 B fragments before the WMMAs -> overlapped ds waits
        v16bf bfr[8];
        const int kb = (lane < 16) ? 0 : 8;
        #pragma unroll
        for (int j = 0; j < 8; ++j) {
            const int col = (j << 4) + (lane & 15);
            *(u32v4*)&bfr[j]       = *(const u32v4*)&Bs[buf][col][kb];
            *((u32v4*)&bfr[j] + 1) = *(const u32v4*)&Bs[buf][col][kb + 16];
        }
        #pragma unroll
        for (int j = 0; j < 8; ++j) {
            acc[j] = __builtin_amdgcn_wmma_f32_16x16x32_bf16(
                false, afrag, false, bfr[j], (short)0, acc[j], false, false);
        }

        if (issuer && more) __builtin_amdgcn_s_wait_tensorcnt(0);
        __syncthreads();
        buf ^= 1;
    }

    // C layout: lane<16 -> M=r, N=lane ; lane>=16 -> M=r+8, N=lane-16
    // x_proj is a 512MB single-use stream: keep it out of L2 (nontemporal).
    for (int j = 0; j < 8; ++j) {
        const size_t col = bcol0 + (j << 4) + (lane & 15);
        for (int r = 0; r < 8; ++r) {
            const size_t row = arow0 + (wave << 4) + ((lane < 16) ? r : r + 8);
            __builtin_nontemporal_store(acc[j][r], &C[row * FOURU + col]);
        }
    }
}

// ---------------------------------------------------------------------------
// Persistent recurrent kernel: 64 WGs x 256 threads. WG w owns u-slice
// [w*16, w*16+16) and its four gate column blocks {w, w+64, w+128, w+192}
// (16 wide each) of z = h @ Wh. c-state lives in LDS for all 512 steps.
// Wh^T (8MB bf16) stays L2-resident; x_proj read nontemporally (single use).
// ---------------------------------------------------------------------------
__global__ __launch_bounds__(256) void lstm_recurrent_kernel(
    const float* __restrict__ xproj,  // [BT][4096], row = b*T + t
    const bf16*  __restrict__ Wht,    // [4096][1024] (Wh transposed)
    const float* __restrict__ bias,   // [4096]
    bf16*        __restrict__ hbuf,   // [64][1024] bf16 h state
    float*       __restrict__ out,    // [B][T][U]
    unsigned int* __restrict__ barrier_ctr)
{
    __shared__ __align__(16) float zbuf[64][64];  // [batch][gate*16 + u]
    __shared__ __align__(16) float cbuf[64][16];  // cell state slice

    const int wave = threadIdx.x >> 5;
    const int lane = threadIdx.x & 31;
    const int u0   = blockIdx.x * 16;

    for (int e = threadIdx.x; e < 64 * 16; e += 256)
        cbuf[e >> 4][e & 15] = 0.0f;
    __syncthreads();

    for (int t = 0; t < T_; ++t) {
        // ---- z = h @ Wh for this WG's 64x64 block (4 m-tiles x 4 gates) ----
        for (int s = 0; s < 2; ++s) {
            const int tid2 = wave * 2 + s;     // 0..15
            const int m    = tid2 >> 2;        // batch tile 0..3
            const int g    = tid2 & 3;         // gate 0..3
            const int ncol0 = g * 1024 + u0;   // column base in z [0,4096)

            v8f acc = zero8();
            const int arow   = (m << 4) + (lane & 15);
            const int kboffA = (lane < 16) ? 0 : 8;
            const int bcol   = ncol0 + (lane & 15);
            const int kboffB = (lane < 16) ? 0 : 8;
            const bf16* Aptr = hbuf + (size_t)arow * U_;
            const bf16* Bptr = Wht  + (size_t)bcol * U_;

            // unroll by 2: 8 global loads in flight before first WMMA wait
            for (int kk = 0; kk < U_; kk += 64) {
                v16bf a0, b0, a1, b1;
                *(u32v4*)&a0       = *(const u32v4*)(Aptr + kk + kboffA);
                *((u32v4*)&a0 + 1) = *(const u32v4*)(Aptr + kk + kboffA + 16);
                *(u32v4*)&b0       = *(const u32v4*)(Bptr + kk + kboffB);
                *((u32v4*)&b0 + 1) = *(const u32v4*)(Bptr + kk + kboffB + 16);
                *(u32v4*)&a1       = *(const u32v4*)(Aptr + kk + 32 + kboffA);
                *((u32v4*)&a1 + 1) = *(const u32v4*)(Aptr + kk + 32 + kboffA + 16);
                *(u32v4*)&b1       = *(const u32v4*)(Bptr + kk + 32 + kboffB);
                *((u32v4*)&b1 + 1) = *(const u32v4*)(Bptr + kk + 32 + kboffB + 16);
                acc = __builtin_amdgcn_wmma_f32_16x16x32_bf16(
                    false, a0, false, b0, (short)0, acc, false, false);
                acc = __builtin_amdgcn_wmma_f32_16x16x32_bf16(
                    false, a1, false, b1, (short)0, acc, false, false);
            }
            for (int r = 0; r < 8; ++r) {
                const int row = (m << 4) + ((lane < 16) ? r : r + 8);
                zbuf[row][(g << 4) + (lane & 15)] = acc[r];
            }
        }
        __syncthreads();

        // ---- gates: 64 batch x 16 u = 1024 elements, 4 per thread ----
        for (int e = threadIdx.x; e < 1024; e += 256) {
            const int brow = e >> 4;
            const int uc   = e & 15;
            const size_t xrow = (size_t)brow * T_ + t;
            const float* xp = &xproj[xrow * FOURU + u0 + uc];
            const float zi = zbuf[brow][uc]      + __builtin_nontemporal_load(xp +    0) + bias[       u0 + uc];
            const float zf = zbuf[brow][16 + uc] + __builtin_nontemporal_load(xp + 1024) + bias[1024 + u0 + uc];
            const float zg = zbuf[brow][32 + uc] + __builtin_nontemporal_load(xp + 2048) + bias[2048 + u0 + uc];
            const float zo = zbuf[brow][48 + uc] + __builtin_nontemporal_load(xp + 3072) + bias[3072 + u0 + uc];

            const float ig = 1.0f / (1.0f + __expf(-zi));
            const float fg = 1.0f / (1.0f + __expf(-zf));
            const float og = 1.0f / (1.0f + __expf(-zo));
            const float c  = fg * cbuf[brow][uc] + ig * tanhf(zg);
            cbuf[brow][uc] = c;
            const float h  = og * tanhf(c);

            __builtin_nontemporal_store(h, &out[xrow * U_ + u0 + uc]);
            hbuf[(size_t)brow * U_ + u0 + uc] = (bf16)h;
        }

        // ---- device-wide barrier: all h writes visible before next step ----
        __threadfence();
        __syncthreads();
        if (threadIdx.x == 0) {
            atomicAdd(barrier_ctr, 1u);
            const unsigned int target = 64u * (unsigned int)(t + 1);
            while (*(volatile unsigned int*)barrier_ctr < target) {
                __builtin_amdgcn_s_sleep(1);
            }
            __threadfence();
        }
        __syncthreads();
    }
}

// ---------------------------------------------------------------------------
extern "C" void kernel_launch(void* const* d_in, const int* in_sizes, int n_in,
                              void* d_out, int out_size, void* d_ws, size_t ws_size,
                              hipStream_t stream)
{
    const float* x    = (const float*)d_in[0];   // [B,T,D]
    const float* Wx   = (const float*)d_in[1];   // [D,4U]
    const float* Wh   = (const float*)d_in[2];   // [U,4U]
    const float* bias = (const float*)d_in[3];   // [4U]
    float* out = (float*)d_out;                  // [B,T,U]

    // workspace carving
    char* ws = (char*)d_ws;
    const size_t xbf_bytes   = (size_t)BT * D_ * sizeof(bf16);      // 64 MB
    const size_t wt_bytes    = (size_t)D_ * FOURU * sizeof(bf16);   // 8 MB each
    const size_t xproj_bytes = (size_t)BT * FOURU * sizeof(float);  // 512 MB

    bf16*  x_bf  = (bf16*)ws;
    bf16*  Wxt   = (bf16*)(ws + xbf_bytes);
    bf16*  Wht   = (bf16*)(ws + xbf_bytes + wt_bytes);
    float* xproj = (float*)(ws + xbf_bytes + 2 * wt_bytes);
    bf16*  hbuf  = (bf16*)(ws + xbf_bytes + 2 * wt_bytes + xproj_bytes);
    unsigned int* ctr = (unsigned int*)((char*)hbuf + (size_t)B_ * U_ * sizeof(bf16));

    // Phase 1: precision conversion / weight transposition
    f32_to_bf16_kernel<<<(BT * D_ + 255) / 256, 256, 0, stream>>>(x, x_bf, BT * D_);
    transpose_to_bf16_kernel<<<dim3(FOURU / 32, D_ / 32), 256, 0, stream>>>(Wx, Wxt, D_, FOURU);
    transpose_to_bf16_kernel<<<dim3(FOURU / 32, U_ / 32), 256, 0, stream>>>(Wh, Wht, U_, FOURU);
    init_state_kernel<<<(B_ * U_ + 255) / 256, 256, 0, stream>>>(hbuf, ctr, B_ * U_);

    // Phase 2: big input-projection GEMM (compute bound, WMMA bf16 + TDM)
    xproj_gemm_kernel<<<dim3(FOURU / 128, BT / 128), 256, 0, stream>>>(x_bf, Wxt, xproj);

    // Phase 3: persistent recurrence, one grid barrier per timestep
    lstm_recurrent_kernel<<<64, 256, 0, stream>>>(xproj, Wht, bias, hbuf, out, ctr);
}
